// CrossAttention_42975442764662
// MI455X (gfx1250) — matrix operand order repaired and checked by reading, not compiled, not run on hardware
//
#include <hip/hip_runtime.h>
#include <hip/hip_bf16.h>

typedef __attribute__((ext_vector_type(16))) _Float16 v16h;
typedef __attribute__((ext_vector_type(8)))  _Float16 v8h;
typedef __attribute__((ext_vector_type(4)))  _Float16 v4h;
typedef __attribute__((ext_vector_type(8)))  float    v8f;

#define B_  4
#define N_  4096
#define C1_ 320
#define C2_ 256
#define H_  5
#define HD_ 64

// ---------------- fragment loaders (wave32 WMMA operands, contiguous k order) ----------------
// A 16x32 f16: lane L: row = L&15, halves i = A[row, kb+i],
//              kb = (L>>4)*16
__device__ __forceinline__ v16h ld_frag_a(const _Float16* p) {
  v8h lo = *(const v8h*)(p);
  v8h hi = *(const v8h*)(p + 8);
  v16h r;
#pragma unroll
  for (int i = 0; i < 8; ++i) { r[i] = lo[i]; r[i + 8] = hi[i]; }
  return r;
}
// B 32x16 f16: lane L: col = L&15, halves i = B[kb+i, col], kb = (L>>4)*16
// (16 contiguous halves when B is stored K-contiguous per column)
__device__ __forceinline__ v16h ld_frag_b(const _Float16* p) {
  v8h lo = *(const v8h*)(p);
  v8h hi = *(const v8h*)(p + 8);
  v16h r;
#pragma unroll
  for (int i = 0; i < 8; ++i) { r[i] = lo[i]; r[i + 8] = hi[i]; }
  return r;
}

__device__ __forceinline__ v8f wmma_f16(v16h a, v16h b, v8f c) {
  return __builtin_amdgcn_wmma_f32_16x16x32_f16(false, a, false, b, (short)0, c,
                                                false, false);
}

// One wave computes a 32x64 block of OUT = A[32,K] @ W^T (W row-major [Nout,K]).
// 8 WMMAs per k-step from 2 A-frags + 4 B-frags (weight frags reused across the
// two row-tiles -> higher wmma:vmem ratio).
__device__ __forceinline__ void gemm8(const _Float16* __restrict__ Ap, int lda,
                                      const _Float16* __restrict__ Wp, int ldw,
                                      int K, v8f c[2][4]) {
  const int lane = threadIdx.x & 31;
  const _Float16* a0_ptr = Ap + (lane & 15) * lda + ((lane >> 4) * 16);
  const _Float16* a1_ptr = a0_ptr + 16 * lda;
  const _Float16* b_ptr  = Wp + (lane & 15) * ldw + ((lane >> 4) * 16);
  for (int k0 = 0; k0 < K; k0 += 32) {
    v16h a0 = ld_frag_a(a0_ptr + k0);
    v16h a1 = ld_frag_a(a1_ptr + k0);
#pragma unroll
    for (int j = 0; j < 4; ++j) {
      v16h b = ld_frag_b(b_ptr + j * 16 * ldw + k0);
      c[0][j] = wmma_f16(a0, b, c[0][j]);
      c[1][j] = wmma_f16(a1, b, c[1][j]);
    }
  }
}

// ---------------- f32 -> f16 (x4 vectorized) ----------------
__global__ void k_cvt(const float* __restrict__ s, _Float16* __restrict__ d, int n) {
  int i = (blockIdx.x * blockDim.x + threadIdx.x) * 4;
  if (i < n) {
    float4 v = *(const float4*)(s + i);
    v4h o = {(_Float16)v.x, (_Float16)v.y, (_Float16)v.z, (_Float16)v.w};
    *(v4h*)(d + i) = o;
  }
}

// ---------------- Q projection: Q[b,h,n,hd] = qscale * (x @ qw^T) -------------
// qscale folds HD^-0.5 * log2(e) so attention softmax runs in exp2 domain.
__global__ void __launch_bounds__(128) k_gemm_q(const _Float16* __restrict__ A,
                                                const _Float16* __restrict__ W,
                                                int K, _Float16* __restrict__ Qb,
                                                float qscale) {
  const int wave = (blockIdx.x * 128 + (int)threadIdx.x) >> 5;
  const int NG = C1_ / 64;  // 5 strips of 64 cols
  const int mt = wave / NG;
  const int ng = wave % NG;
  v8f c[2][4] = {};
  gemm8(A + (size_t)mt * 32 * K, K, W + (size_t)ng * 64 * K, K, K, c);
  const int lane = threadIdx.x & 31;
  const int col = lane & 15, rb = (lane >> 4) * 8;
#pragma unroll
  for (int mi = 0; mi < 2; ++mi) {
#pragma unroll
    for (int j = 0; j < 4; ++j) {
      const int nout0 = ng * 64 + j * 16;
#pragma unroll
      for (int i = 0; i < 8; ++i) {
        int m = mt * 32 + mi * 16 + rb + i;
        int b = m >> 12, n = m & (N_ - 1);
        int nout = nout0 + col, h = nout >> 6, hd = nout & 63;
        Qb[(((size_t)(b * H_ + h)) * N_ + n) * HD_ + hd] =
            (_Float16)(c[mi][j][i] * qscale);
      }
    }
  }
}

// ---------------- KV projection: K->[B,H,N,HD], V->[B,H,HD,N] (transposed) ----
__global__ void __launch_bounds__(128) k_gemm_kv(const _Float16* __restrict__ A,
                                                 const _Float16* __restrict__ W,
                                                 int K, _Float16* __restrict__ Kb,
                                                 _Float16* __restrict__ Vt) {
  const int wave = (blockIdx.x * 128 + (int)threadIdx.x) >> 5;
  const int NG = (2 * C1_) / 64;  // 10
  const int mt = wave / NG;
  const int ng = wave % NG;
  v8f c[2][4] = {};
  gemm8(A + (size_t)mt * 32 * K, K, W + (size_t)ng * 64 * K, K, K, c);
  const int lane = threadIdx.x & 31;
  const int col = lane & 15, rb = (lane >> 4) * 8;
#pragma unroll
  for (int mi = 0; mi < 2; ++mi) {
#pragma unroll
    for (int j = 0; j < 4; ++j) {
      const int nout0 = ng * 64 + j * 16;
#pragma unroll
      for (int i = 0; i < 8; ++i) {
        int m = mt * 32 + mi * 16 + rb + i;
        int b = m >> 12, n = m & (N_ - 1);
        int nout = nout0 + col;
        if (nout < C1_) {
          int h = nout >> 6, hd = nout & 63;
          Kb[(((size_t)(b * H_ + h)) * N_ + n) * HD_ + hd] = (_Float16)c[mi][j][i];
        } else {
          int rem = nout - C1_;
          int h = rem >> 6, hd = rem & 63;
          Vt[(((size_t)(b * H_ + h)) * HD_ + hd) * N_ + n] = (_Float16)c[mi][j][i];
        }
      }
    }
  }
}

// ---------------- cross-lane row reductions (rows live across 16 lanes) -------
__device__ __forceinline__ float redmax16(float v) {
#pragma unroll
  for (int s = 1; s < 16; s <<= 1) v = fmaxf(v, __shfl_xor(v, s, 32));
  return v;
}
__device__ __forceinline__ float redsum16(float v) {
#pragma unroll
  for (int s = 1; s < 16; s <<= 1) v += __shfl_xor(v, s, 32);
  return v;
}

// ---------------- flash attention: one wave = 16 queries of one (b,h) ---------
// 64 keys per block: 16 WMMAs per block, softmax/rescale amortized 2x vs 32-key
// blocks. Probabilities are already exp2-domain (Q pre-scaled by log2 e).
__global__ void __launch_bounds__(128) k_attn(const _Float16* __restrict__ Qb,
                                              const _Float16* __restrict__ Kb,
                                              const _Float16* __restrict__ Vt,
                                              _Float16* __restrict__ Yb) {
  __shared__ __align__(32) _Float16 Plds[4][16 * 64];
  const int wv = threadIdx.x >> 5;
  const int lane = threadIdx.x & 31;
  const int gwave = blockIdx.x * 4 + wv;
  const int qt = gwave & 255;  // N_/16 = 256 q-tiles
  const int bh = gwave >> 8;   // 0..19
  const _Float16* Qp = Qb + ((size_t)bh * N_ + qt * 16) * HD_;
  const _Float16* Kp = Kb + (size_t)bh * N_ * HD_;
  const _Float16* Vp = Vt + (size_t)bh * HD_ * N_;
  const int arow = lane & 15, akb = (lane >> 4) * 16;
  const int col = lane & 15, bkb = (lane >> 4) * 16;
  v16h aq0 = ld_frag_a(Qp + arow * HD_ + akb);       // k 0..31
  v16h aq1 = ld_frag_a(Qp + arow * HD_ + 32 + akb);  // k 32..63
  float mrow[8], lrow[8];
  v8f o[4] = {};
#pragma unroll
  for (int i = 0; i < 8; ++i) { mrow[i] = -3.0e38f; lrow[i] = 0.f; }
  _Float16* P = &Plds[wv][0];

  for (int kt = 0; kt < N_; kt += 64) {
    const _Float16* kp = Kp + (size_t)kt * HD_;
    __builtin_prefetch(kp + 64 * HD_, 0, 0);  // global_prefetch_b8: next K block
    // S = Q @ K^T for 64 keys (B[k,n] = Key[n][k], contiguous per key row)
    v8f s[4] = {};
#pragma unroll
    for (int g = 0; g < 4; ++g) {
      v16h b0 = ld_frag_b(kp + (g * 16 + col) * HD_ + bkb);
      v16h b1 = ld_frag_b(kp + (g * 16 + col) * HD_ + 32 + bkb);
      s[g] = wmma_f16(aq0, b0, s[g]);
      s[g] = wmma_f16(aq1, b1, s[g]);
    }
    // online softmax (exp2 domain); probabilities into LDS row-major [16][64]
#pragma unroll
    for (int i = 0; i < 8; ++i) {
      float mx = fmaxf(fmaxf(s[0][i], s[1][i]), fmaxf(s[2][i], s[3][i]));
      mx = redmax16(mx);
      float mtot = fmaxf(mrow[i], mx);
      float alpha = __builtin_amdgcn_exp2f(mrow[i] - mtot);
      mrow[i] = mtot;
      float p0 = __builtin_amdgcn_exp2f(s[0][i] - mtot);
      float p1 = __builtin_amdgcn_exp2f(s[1][i] - mtot);
      float p2 = __builtin_amdgcn_exp2f(s[2][i] - mtot);
      float p3 = __builtin_amdgcn_exp2f(s[3][i] - mtot);
      lrow[i] = lrow[i] * alpha + redsum16((p0 + p1) + (p2 + p3));
      o[0][i] *= alpha; o[1][i] *= alpha; o[2][i] *= alpha; o[3][i] *= alpha;
      const int r = (lane >> 4) * 8 + i;  // C-layout row for this vgpr slot
      P[r * 64 + col]      = (_Float16)p0;
      P[r * 64 + 16 + col] = (_Float16)p1;
      P[r * 64 + 32 + col] = (_Float16)p2;
      P[r * 64 + 48 + col] = (_Float16)p3;
    }
    asm volatile("s_wait_dscnt 0x0" ::: "memory");
    // P as A-fragments (lda = 64), V^T as B-fragments (contiguous in key dim)
    v16h ap0 = ld_frag_a(P + arow * 64 + akb);       // keys 0..31
    v16h ap1 = ld_frag_a(P + arow * 64 + 32 + akb);  // keys 32..63
    const _Float16* vp = Vp + kt;
#pragma unroll
    for (int j = 0; j < 4; ++j) {
      v16h b0 = ld_frag_b(vp + (size_t)(j * 16 + col) * N_ + bkb);
      v16h b1 = ld_frag_b(vp + (size_t)(j * 16 + col) * N_ + 32 + bkb);
      o[j] = wmma_f16(ap0, b0, o[j]);
      o[j] = wmma_f16(ap1, b1, o[j]);
    }
  }
  // normalize and store y[b, n, h*64 + c] as f16 for the projection GEMM
  const int b = bh / H_, h = bh % H_;
#pragma unroll
  for (int i = 0; i < 8; ++i) {
    const float inv = 1.0f / lrow[i];
    const int n = qt * 16 + ((lane >> 4) * 8) + i;
    _Float16* yp = Yb + ((size_t)b * N_ + n) * C1_ + h * HD_ + col;
    yp[0]  = (_Float16)(o[0][i] * inv);
    yp[16] = (_Float16)(o[1][i] * inv);
    yp[32] = (_Float16)(o[2][i] * inv);
    yp[48] = (_Float16)(o[3][i] * inv);
  }
}

// ---------------- output projection: f32 out into concat half ----------------
__global__ void __launch_bounds__(128) k_gemm_proj(const _Float16* __restrict__ A,
                                                   const _Float16* __restrict__ W,
                                                   const float* __restrict__ bias,
                                                   float* __restrict__ out,
                                                   int halfsel) {
  const int wave = (blockIdx.x * 128 + (int)threadIdx.x) >> 5;
  const int NG = C1_ / 64;  // 5
  const int mt = wave / NG;
  const int ng = wave % NG;
  v8f c[2][4] = {};
  gemm8(A + (size_t)mt * 32 * C1_, C1_, W + (size_t)ng * 64 * C1_, C1_, C1_, c);
  const int lane = threadIdx.x & 31;
  const int col = lane & 15, rb = (lane >> 4) * 8;
#pragma unroll
  for (int mi = 0; mi < 2; ++mi) {
#pragma unroll
    for (int j = 0; j < 4; ++j) {
      const int nout0 = ng * 64 + j * 16;
#pragma unroll
      for (int i = 0; i < 8; ++i) {
        int m = mt * 32 + mi * 16 + rb + i;
        int nout = nout0 + col;
        out[(size_t)m * (2 * C1_) + (size_t)halfsel * C1_ + nout] =
            c[mi][j][i] + bias[nout];
      }
    }
  }
}

extern "C" void kernel_launch(void* const* d_in, const int* in_sizes, int n_in,
                              void* d_out, int out_size, void* d_ws, size_t ws_size,
                              hipStream_t stream) {
  const float* x1      = (const float*)d_in[0];
  const float* x2      = (const float*)d_in[1];
  const float* q1_w    = (const float*)d_in[2];
  const float* q2_w    = (const float*)d_in[3];
  const float* kv1_w   = (const float*)d_in[4];
  const float* kv2_w   = (const float*)d_in[5];
  const float* proj1_w = (const float*)d_in[6];
  const float* proj1_b = (const float*)d_in[7];
  const float* proj2_w = (const float*)d_in[8];
  const float* proj2_b = (const float*)d_in[9];
  float* out = (float*)d_out;

  char* ws = (char*)d_ws;
  size_t off = 0;
  auto alloc = [&](size_t elems) -> _Float16* {
    _Float16* p = (_Float16*)(ws + off);
    off = (off + elems * sizeof(_Float16) + 255) & ~(size_t)255;
    return p;
  };
  const size_t MN = (size_t)B_ * N_;  // 16384 rows
  _Float16* x1h = alloc(MN * C1_);
  _Float16* x2h = alloc(MN * C2_);
  _Float16* q1h = alloc(MN * C1_);
  _Float16* k1h = alloc(MN * C1_);
  _Float16* v1t = alloc(MN * C1_);
  _Float16* q2h = alloc(MN * C1_);
  _Float16* k2h = alloc(MN * C1_);
  _Float16* v2t = alloc(MN * C1_);
  _Float16* y1h = alloc(MN * C1_);
  _Float16* y2h = alloc(MN * C1_);
  _Float16* wq1 = alloc((size_t)C1_ * C1_);
  _Float16* wq2 = alloc((size_t)C1_ * C2_);
  _Float16* wk1 = alloc((size_t)2 * C1_ * C2_);
  _Float16* wk2 = alloc((size_t)2 * C1_ * C1_);
  _Float16* wp1 = alloc((size_t)C1_ * C1_);
  _Float16* wp2 = alloc((size_t)C1_ * C1_);

  auto cvt = [&](const float* s, _Float16* d, size_t n) {
    k_cvt<<<(unsigned)((n / 4 + 255) / 256), 256, 0, stream>>>(s, d, (int)n);
  };
  cvt(x1, x1h, MN * C1_);
  cvt(x2, x2h, MN * C2_);
  cvt(q1_w, wq1, (size_t)C1_ * C1_);
  cvt(q2_w, wq2, (size_t)C1_ * C2_);
  cvt(kv1_w, wk1, (size_t)2 * C1_ * C2_);
  cvt(kv2_w, wk2, (size_t)2 * C1_ * C1_);
  cvt(proj1_w, wp1, (size_t)C1_ * C1_);
  cvt(proj2_w, wp2, (size_t)C1_ * C1_);

  // HD^-0.5 * log2(e): softmax runs in exp2 domain (native v_exp_f32)
  const float qscale = 0.125f * 1.44269504088896340736f;

  // M strips of 32 rows: 512; waves = 512 * (Nout/64); 4 waves/block
  k_gemm_q<<<512 * 5 / 4, 128, 0, stream>>>(x1h, wq1, C1_, q1h, qscale);   // br1 Q
  k_gemm_kv<<<512 * 10 / 4, 128, 0, stream>>>(x2h, wk1, C2_, k1h, v1t);    // br1 KV
  k_gemm_q<<<512 * 5 / 4, 128, 0, stream>>>(x2h, wq2, C2_, q2h, qscale);   // br2 Q
  k_gemm_kv<<<512 * 10 / 4, 128, 0, stream>>>(x1h, wk2, C1_, k2h, v2t);    // br2 KV

  // waves = B*H*(N/16) = 5120; 4 waves/block
  k_attn<<<5120 / 4, 128, 0, stream>>>(q1h, k1h, v1t, y1h);
  k_attn<<<5120 / 4, 128, 0, stream>>>(q2h, k2h, v2t, y2h);

  k_gemm_proj<<<512 * 5 / 4, 128, 0, stream>>>(y1h, wp1, proj1_b, out, 0);
  k_gemm_proj<<<512 * 5 / 4, 128, 0, stream>>>(y2h, wp2, proj2_b, out, 1);
}
